// GMM_73340861546919
// MI455X (gfx1250) — matrix operand order repaired and checked by reading, not compiled, not run on hardware
//
#include <hip/hip_runtime.h>

typedef __attribute__((ext_vector_type(4)))  _Float16 v4h;
typedef __attribute__((ext_vector_type(8)))  _Float16 v8h;
typedef __attribute__((ext_vector_type(16))) _Float16 v16h;
typedef __attribute__((ext_vector_type(4)))  float    v4f;
typedef __attribute__((ext_vector_type(8)))  float    v8f;

#ifndef GMM_M
#define GMM_M 16384
#define GMM_N 8192
#define GMM_D 128
#endif

#if __has_builtin(__builtin_amdgcn_exp2f)
#define EXP2F(x) __builtin_amdgcn_exp2f(x)
#else
#define EXP2F(x) exp2f(x)
#endif

// -------- pre-pass 1: split fp32 -> f16 hi + f16 lo (hi=(f16)x, lo=(f16)(x-hi)) --------
__global__ __launch_bounds__(256) void gmm_split_f16(const float* __restrict__ src,
                                                     _Float16* __restrict__ hi,
                                                     _Float16* __restrict__ lo,
                                                     int n4) {
  int i = blockIdx.x * blockDim.x + threadIdx.x;
  if (i >= n4) return;
  v4f x = ((const v4f*)src)[i];
  v4h h, l;
#pragma unroll
  for (int k = 0; k < 4; ++k) {
    _Float16 hk = (_Float16)x[k];
    h[k] = hk;
    l[k] = (_Float16)(x[k] - (float)hk);
  }
  ((v4h*)hi)[i] = h;
  ((v4h*)lo)[i] = l;
}

// -------- pre-pass 2: per-row squared L2 norm (one wave32 per row, D=128) --------
__global__ __launch_bounds__(256) void gmm_row_sqnorm(const float* __restrict__ src,
                                                      float* __restrict__ dst,
                                                      int rows) {
  int wave = (int)((blockIdx.x * blockDim.x + threadIdx.x) >> 5);
  int lane = threadIdx.x & 31;
  if (wave >= rows) return;
  const float* r = src + (size_t)wave * GMM_D;
  float s = 0.f;
#pragma unroll
  for (int k = 0; k < GMM_D; k += 32) {
    float v = r[k + lane];
    s = fmaf(v, v, s);
  }
#pragma unroll
  for (int m = 16; m >= 1; m >>= 1) s += __shfl_xor(s, m, 32);
  if (lane == 0) dst[wave] = s;
}

// -------- main kernel: f16x3 WMMA gram tiles + fused gaussian-kernel reduction --------
// block = 256 threads = 8 waves; each block owns a 16-row slab of x,
// each wave streams N-tiles (stride 8) and keeps a private weighted accumulator.
__global__ __launch_bounds__(256) void gmm_wmma_main(
    const _Float16* __restrict__ xh, const _Float16* __restrict__ xl,
    const _Float16* __restrict__ mh, const _Float16* __restrict__ ml,
    const float* __restrict__ x2, const float* __restrict__ m2,
    const float* __restrict__ w, float* __restrict__ out) {
  __shared__ float part[8][16];

  const int lane = threadIdx.x & 31;
  const int wv   = threadIdx.x >> 5;
  const int i0   = blockIdx.x * 16;
  const int h    = lane >> 4;   // half-wave select (0: lanes 0-15, 1: lanes 16-31)
  const int lm   = lane & 15;
  const int arow = i0 + lm;     // A-matrix row owned by this lane

  // ---- load A (x rows) once: 16x32 A tile in contiguous K order (same as B), 4 K-chunks, hi & lo ----
  // lanes 0-15:  vgpr0-7 = K 0..15   (of each 32-chunk)
  // lanes 16-31: vgpr0-7 = K 16..31
  v16h ahv[4], alv[4];
  {
    const _Float16* xr = xh + (size_t)arow * GMM_D;
    const _Float16* xs = xl + (size_t)arow * GMM_D;
#pragma unroll
    for (int c = 0; c < 4; ++c) {
      v8h p0 = *(const v8h*)(xr + 32 * c + 16 * h);
      v8h p1 = *(const v8h*)(xr + 32 * c + 16 * h + 8);
      ahv[c] = __builtin_shufflevector(p0, p1, 0, 1, 2, 3, 4, 5, 6, 7, 8, 9, 10, 11, 12, 13, 14, 15);
      v8h q0 = *(const v8h*)(xs + 32 * c + 16 * h);
      v8h q1 = *(const v8h*)(xs + 32 * c + 16 * h + 8);
      alv[c] = __builtin_shufflevector(q0, q1, 0, 1, 2, 3, 4, 5, 6, 7, 8, 9, 10, 11, 12, 13, 14, 15);
    }
  }

  const float NL2E = -0.7213475204444817f;  // -log2(e)/2  (sigma = 1)
  const float L2E  =  1.4426950408889634f;  //  log2(e)    (multiplies dot: -2*dot*NL2E)

  // per-lane row-dependent constant:  -log2(e)/2 * |x_i|^2
  float x2n[8];
#pragma unroll
  for (int v = 0; v < 8; ++v) x2n[v] = NL2E * x2[i0 + v + 8 * h];

  v8f acc = {};  // Sum_j w[j]*exp2(t) for this lane's columns

  for (int jt = wv; jt < GMM_N / 16; jt += 8) {
    const int j = jt * 16 + lm;  // B column owned by this lane
    // B 32x16 in contiguous K order (same as A): lane holds 16 contiguous K values starting at kc + 16*h
    const _Float16* bhp = mh + (size_t)j * GMM_D + 16 * h;
    const _Float16* blp = ml + (size_t)j * GMM_D + 16 * h;

    v8f c4 = {};
#pragma unroll
    for (int c = 0; c < 4; ++c) {
      v16h bhv = *(const v16h*)(bhp + 32 * c);
      v16h blv = *(const v16h*)(blp + 32 * c);
      c4 = __builtin_amdgcn_wmma_f32_16x16x32_f16(false, ahv[c], false, bhv, (short)0, c4, false, false);
      c4 = __builtin_amdgcn_wmma_f32_16x16x32_f16(false, ahv[c], false, blv, (short)0, c4, false, false);
      c4 = __builtin_amdgcn_wmma_f32_16x16x32_f16(false, alv[c], false, bhv, (short)0, c4, false, false);
    }

    const float m2j = m2[j];
    const float wj  = w[j];
#pragma unroll
    for (int v = 0; v < 8; ++v) {
      // t = -log2(e)/2 * (|x|^2 + |m|^2 - 2*dot)
      float bx = fmaf(m2j, NL2E, x2n[v]);
      float t  = fmaf(c4[v], L2E, bx);
      t = fminf(t, 0.0f);                 // sq clamp at 0  ->  t clamp at 0
      acc[v] = fmaf(EXP2F(t), wj, acc[v]);
    }
  }

  // reduce over the 16 lanes of each half-wave (columns), per C/D row (vgpr)
#pragma unroll
  for (int v = 0; v < 8; ++v) {
#pragma unroll
    for (int m = 8; m >= 1; m >>= 1) acc[v] += __shfl_xor(acc[v], m, 32);
  }
  if (lm == 0) {
#pragma unroll
    for (int v = 0; v < 8; ++v) part[wv][v + 8 * h] = acc[v];
  }
  __syncthreads();

  if (threadIdx.x < 16) {
    float s = 0.f;
#pragma unroll
    for (int q = 0; q < 8; ++q) s += part[q][threadIdx.x];
    out[i0 + threadIdx.x] = 0.15915494309189535f * s;  // 1/(2*pi), sigma=1
  }
}

// ---------------------------------------------------------------------------
extern "C" void kernel_launch(void* const* d_in, const int* in_sizes, int n_in,
                              void* d_out, int out_size, void* d_ws, size_t ws_size,
                              hipStream_t stream) {
  (void)in_sizes; (void)n_in; (void)out_size; (void)ws_size;
  const float* x  = (const float*)d_in[0];   // (M, D) fp32
  const float* mu = (const float*)d_in[1];   // (N, D) fp32
  const float* wt = (const float*)d_in[2];   // (N,)   fp32
  float* out = (float*)d_out;                // (M,)   fp32

  // workspace layout
  char* ws = (char*)d_ws;
  size_t off = 0;
  _Float16* xh = (_Float16*)(ws + off); off += (size_t)GMM_M * GMM_D * sizeof(_Float16);
  _Float16* xl = (_Float16*)(ws + off); off += (size_t)GMM_M * GMM_D * sizeof(_Float16);
  _Float16* mh = (_Float16*)(ws + off); off += (size_t)GMM_N * GMM_D * sizeof(_Float16);
  _Float16* ml = (_Float16*)(ws + off); off += (size_t)GMM_N * GMM_D * sizeof(_Float16);
  float*    x2 = (float*)(ws + off);    off += (size_t)GMM_M * sizeof(float);
  float*    m2 = (float*)(ws + off);    off += (size_t)GMM_N * sizeof(float);

  // pre-pass: fp32 -> f16 hi/lo splits
  {
    int n4 = GMM_M * GMM_D / 4;
    gmm_split_f16<<<(n4 + 255) / 256, 256, 0, stream>>>(x, xh, xl, n4);
  }
  {
    int n4 = GMM_N * GMM_D / 4;
    gmm_split_f16<<<(n4 + 255) / 256, 256, 0, stream>>>(mu, mh, ml, n4);
  }
  // pre-pass: squared norms (8 rows per 256-thread block)
  gmm_row_sqnorm<<<GMM_M / 8, 256, 0, stream>>>(x, x2, GMM_M);
  gmm_row_sqnorm<<<GMM_N / 8, 256, 0, stream>>>(mu, m2, GMM_N);

  // main fused WMMA kernel: one block per 16-row slab
  gmm_wmma_main<<<GMM_M / 16, 256, 0, stream>>>(xh, xl, mh, ml, x2, m2, wt, out);
}
